// SparseAutoEncoder_74474732913205
// MI455X (gfx1250) — matrix operand order repaired and checked by reading, not compiled, not run on hardware
//
#include <hip/hip_runtime.h>

#define BATCH 16384
#define DIM   1024
#define HID   4096
#define TOPK  64
#define TOPK4 256
#define KAUX  512
#define DEADT 30

typedef __attribute__((ext_vector_type(8)))  float  v8f;
typedef __attribute__((ext_vector_type(4)))  __bf16 v4bf;
typedef __attribute__((ext_vector_type(8)))  __bf16 v8bf;
typedef __attribute__((ext_vector_type(16))) __bf16 v16bf;

// Merge two 16-byte LDS chunks into one WMMA operand (2x ds_load_b128 expected).
static __device__ __forceinline__ v16bf frag2(const __bf16* p0, const __bf16* p1) {
  const v8bf a = *(const v8bf*)p0;
  const v8bf b = *(const v8bf*)p1;
  return __builtin_shufflevector(a, b, 0, 1, 2, 3, 4, 5, 6, 7,
                                       8, 9, 10, 11, 12, 13, 14, 15);
}

// ---------------------------------------------------------------------------
// Split-bf16 (bf16x2 compensated) WMMA GEMM:  C[M,N] = A[M,K] (*) B + cbias[n]
//   B_NK  = true : Btile[n][k] = Bsrc[n*ldb + k]   (encode: W row-major = W^T GEMM)
//   B_NK  = false: Btile[n][k] = Bsrc[k*ldb + n]   (decode: W as [K,N])
//   SUB_ABIAS: A element -> A - abias[k]            (encode: x - pre_bias)
// ---------------------------------------------------------------------------
#define BM 128
#define BN 128
#define BK 32
#define LDT (BK + 8)

template<bool B_NK, bool SUB_ABIAS>
__global__ __launch_bounds__(256)
void gemm_bf16x2(const float* __restrict__ Asrc, int lda,
                 const float* __restrict__ abias,
                 const float* __restrict__ Bsrc, int ldb,
                 const float* __restrict__ cbias,
                 float* __restrict__ Cdst, int ldc, int Ktot)
{
  __shared__ __align__(16) __bf16 AsH[BM * LDT];
  __shared__ __align__(16) __bf16 AsL[BM * LDT];
  __shared__ __align__(16) __bf16 BsH[BN * LDT];
  __shared__ __align__(16) __bf16 BsL[BN * LDT];

  const int t    = threadIdx.x;
  const int lane = t & 31;
  const int wave = t >> 5;       // 0..7
  const int wm   = wave & 1;     // 2 waves along M (64 rows each)
  const int wn   = wave >> 1;    // 4 waves along N (32 cols each)
  const int gm0  = blockIdx.y * BM;
  const int gn0  = blockIdx.x * BN;
  const int half = lane >> 4;    // 0/1 lane-half
  const int l16  = lane & 15;

  v8f acc[4][2];
#pragma unroll
  for (int a = 0; a < 4; ++a)
#pragma unroll
    for (int b = 0; b < 2; ++b)
#pragma unroll
      for (int r = 0; r < 8; ++r) acc[a][b][r] = 0.0f;

  for (int k0 = 0; k0 < Ktot; k0 += BK) {
    // ---- stage A tile (BM x BK): f32 -> (hi,lo) bf16, 8B packed LDS stores
    {
      const int k4 = t & 7;           // 8 float4 chunks = 32 cols
      int m = t >> 3;                 // 32 rows / pass
      if (k0 + BK < Ktot)
        __builtin_prefetch(Asrc + (size_t)(gm0 + m) * lda + (k0 + BK), 0, 0);
#pragma unroll
      for (int rep = 0; rep < 4; ++rep, m += 32) {
        const float4 v = *(const float4*)(Asrc + (size_t)(gm0 + m) * lda + (k0 + k4 * 4));
        float f[4] = {v.x, v.y, v.z, v.w};
        v4bf hi4, lo4;
#pragma unroll
        for (int j = 0; j < 4; ++j) {
          float fv = f[j];
          if (SUB_ABIAS) fv -= abias[k0 + k4 * 4 + j];
          const __bf16 hi = (__bf16)fv;
          hi4[j] = hi;
          lo4[j] = (__bf16)(fv - (float)hi);
        }
        *(v4bf*)(&AsH[m * LDT + k4 * 4]) = hi4;
        *(v4bf*)(&AsL[m * LDT + k4 * 4]) = lo4;
      }
    }
    // ---- stage B tile (BN x BK) in [n][k] layout
    if (B_NK) {
      const int k4 = t & 7;
      int n = t >> 3;
#pragma unroll
      for (int rep = 0; rep < 4; ++rep, n += 32) {
        const float4 v = *(const float4*)(Bsrc + (size_t)(gn0 + n) * ldb + (k0 + k4 * 4));
        float f[4] = {v.x, v.y, v.z, v.w};
        v4bf hi4, lo4;
#pragma unroll
        for (int j = 0; j < 4; ++j) {
          const __bf16 hi = (__bf16)f[j];
          hi4[j] = hi;
          lo4[j] = (__bf16)(f[j] - (float)hi);
        }
        *(v4bf*)(&BsH[n * LDT + k4 * 4]) = hi4;
        *(v4bf*)(&BsL[n * LDT + k4 * 4]) = lo4;
      }
    } else {
      const int n4 = t & 31;          // 32 float4 chunks over 128 cols (coalesced)
      int k = t >> 5;                 // 8 k / pass
#pragma unroll
      for (int rep = 0; rep < 4; ++rep, k += 8) {
        const float4 v = *(const float4*)(Bsrc + (size_t)(k0 + k) * ldb + (gn0 + n4 * 4));
        float f[4] = {v.x, v.y, v.z, v.w};
#pragma unroll
        for (int j = 0; j < 4; ++j) {
          const int n = n4 * 4 + j;
          const __bf16 hi = (__bf16)f[j];
          BsH[n * LDT + k] = hi;
          BsL[n * LDT + k] = (__bf16)(f[j] - (float)hi);
        }
      }
    }
    __syncthreads();

    // ---- build fragments per documented CDNA5 16-bit layouts (2x b128 each)
    v16bf aH[4], aL[4], bH[2], bL[2];
#pragma unroll
    for (int tm = 0; tm < 4; ++tm) {
      const int rb = (wm * 64 + tm * 16 + l16) * LDT + half * 8;
      aH[tm] = frag2(&AsH[rb], &AsH[rb + 16]);   // K = half*8..+7 | 16+half*8..+7
      aL[tm] = frag2(&AsL[rb], &AsL[rb + 16]);
    }
#pragma unroll
    for (int tn = 0; tn < 2; ++tn) {
      const int cb = (wn * 32 + tn * 16 + l16) * LDT + half * 8;
      bH[tn] = frag2(&BsH[cb], &BsH[cb + 16]);   // K = half*8..+7 | 16+half*8..+7
      bL[tn] = frag2(&BsL[cb], &BsL[cb + 16]);
    }

    // ---- 3 WMMAs per tile: hi*hi + hi*lo + lo*hi (compensated bf16)
#pragma unroll
    for (int tm = 0; tm < 4; ++tm)
#pragma unroll
      for (int tn = 0; tn < 2; ++tn) {
        acc[tm][tn] = __builtin_amdgcn_wmma_f32_16x16x32_bf16(
            false, aH[tm], false, bH[tn], (short)0, acc[tm][tn], false, false);
        acc[tm][tn] = __builtin_amdgcn_wmma_f32_16x16x32_bf16(
            false, aH[tm], false, bL[tn], (short)0, acc[tm][tn], false, false);
        acc[tm][tn] = __builtin_amdgcn_wmma_f32_16x16x32_bf16(
            false, aL[tm], false, bH[tn], (short)0, acc[tm][tn], false, false);
      }
    __syncthreads();
  }

  // ---- epilogue: C f32 layout (VGPR r: M=r + 8*half, N=lane%16), + col bias
  const int mofs = half * 8;
#pragma unroll
  for (int tm = 0; tm < 4; ++tm)
#pragma unroll
    for (int tn = 0; tn < 2; ++tn) {
      const int col = gn0 + wn * 32 + tn * 16 + l16;
      const float cb = cbias[col];
#pragma unroll
      for (int r = 0; r < 8; ++r) {
        const int row = gm0 + wm * 64 + tm * 16 + mofs + r;
        Cdst[(size_t)row * ldc + col] = acc[tm][tn][r] + cb;
      }
    }
}

// ---------------------------------------------------------------------------
// Per-row top-K via bitwise radix-select of the K-th largest key in LDS.
// Tie-break: strictly-greater elements first (any order), then ==threshold
// elements in ascending index order (matches jax.lax.top_k selection set).
// ---------------------------------------------------------------------------
template<int KSEL>
__global__ __launch_bounds__(256)
void topk_rows(const float* __restrict__ pre, const int* __restrict__ stats_mask,
               float* __restrict__ vals, int* __restrict__ idxs)
{
  __shared__ unsigned keys[HID];
  __shared__ unsigned cnt;
  __shared__ int sc[256];
  const int row = blockIdx.x;
  const int t = threadIdx.x;
  const float* prow = pre + (size_t)row * HID;

  for (int i = t; i < HID; i += 256) {
    float v = prow[i];
    if (stats_mask) v = (stats_mask[i] > DEADT) ? v : 0.0f;
    const unsigned u = __float_as_uint(v);
    keys[i] = (u & 0x80000000u) ? ~u : (u | 0x80000000u);   // order-preserving
  }
  __syncthreads();

  unsigned prefix = 0u;
  for (int bit = 31; bit >= 0; --bit) {
    const unsigned cand = prefix | (1u << bit);
    if (t == 0) cnt = 0u;
    __syncthreads();
    unsigned c = 0;
    for (int i = t; i < HID; i += 256) c += (keys[i] >= cand) ? 1u : 0u;
    atomicAdd(&cnt, c);
    __syncthreads();
    if (cnt >= (unsigned)KSEL) prefix = cand;
    __syncthreads();
  }
  const unsigned T = prefix;          // K-th largest key

  if (t == 0) cnt = 0u;
  __syncthreads();
  for (int i = t; i < HID; i += 256) {
    if (keys[i] > T) {
      const unsigned p = atomicAdd(&cnt, 1u);
      const unsigned k = keys[i];
      const unsigned u = (k & 0x80000000u) ? (k & 0x7FFFFFFFu) : ~k;
      idxs[(size_t)row * KSEL + p] = i;
      vals[(size_t)row * KSEL + p] = __uint_as_float(u);
    }
  }
  __syncthreads();
  const int n_gt = (int)cnt;
  const int need = KSEL - n_gt;

  const int base = t * (HID / 256);
  int eq = 0;
#pragma unroll
  for (int j = 0; j < HID / 256; ++j) eq += (keys[base + j] == T) ? 1 : 0;
  sc[t] = eq;
  __syncthreads();
  for (int s = 1; s < 256; s <<= 1) {           // Hillis-Steele inclusive scan
    int add = (t >= s) ? sc[t - s] : 0;
    __syncthreads();
    sc[t] += add;
    __syncthreads();
  }
  int off = sc[t] - eq;                         // exclusive prefix (index order)
  const unsigned uT = (T & 0x80000000u) ? (T & 0x7FFFFFFFu) : ~T;
  const float fT = __uint_as_float(uT);
  int r = 0;
  for (int j = 0; j < HID / 256; ++j) {
    if (keys[base + j] == T) {
      const int pos = off + r; ++r;
      if (pos < need) {
        idxs[(size_t)row * KSEL + n_gt + pos] = base + j;
        vals[(size_t)row * KSEL + n_gt + pos] = fT;
      }
    }
  }
}

// ---------------------------------------------------------------------------
// Small elementwise / scatter kernels
// ---------------------------------------------------------------------------
__global__ void ew_sub_colbias(const float* __restrict__ a, const float* __restrict__ bias,
                               float* __restrict__ o, long n, int ld)
{
  const long stride = (long)gridDim.x * blockDim.x;
  for (long i = (long)blockIdx.x * blockDim.x + threadIdx.x; i < n; i += stride)
    o[i] = a[i] - bias[i % ld];
}

__global__ void hits_accum(const float* __restrict__ vals, const int* __restrict__ idxs,
                           long n, int* __restrict__ hits)
{
  const long stride = (long)gridDim.x * blockDim.x;
  for (long i = (long)blockIdx.x * blockDim.x + threadIdx.x; i < n; i += stride)
    if (vals[i] > 1e-5f) atomicAdd(&hits[idxs[i]], 1);
}

__global__ void stats_update(const int* __restrict__ hits, const int* __restrict__ sin,
                             int* __restrict__ sout)
{
  const int h = blockIdx.x * blockDim.x + threadIdx.x;
  if (h < HID) sout[h] = (hits[h] > 0) ? 1 : (sin[h] + 1);
}

__global__ void mask_pre(float* __restrict__ pre, const int* __restrict__ s1,
                         const int* __restrict__ s2, long n)
{
  const long stride = (long)gridDim.x * blockDim.x;
  for (long i = (long)blockIdx.x * blockDim.x + threadIdx.x; i < n; i += stride) {
    const int h = (int)(i % HID);
    const float m = (s1[h] > DEADT && s2[h] > DEADT) ? 1.0f : 0.0f;
    pre[i] *= m;
  }
}

__global__ void scatter_relu(const float* __restrict__ vals, const int* __restrict__ idxs,
                             int ksel, float* __restrict__ lat)
{
  const int row = blockIdx.x;
  for (int j = threadIdx.x; j < ksel; j += blockDim.x) {
    const float v = vals[(size_t)row * ksel + j];
    lat[(size_t)row * HID + idxs[(size_t)row * ksel + j]] = v > 0.0f ? v : 0.0f;
  }
}

// ---------------------------------------------------------------------------
extern "C" void kernel_launch(void* const* d_in, const int* in_sizes, int n_in,
                              void* d_out, int out_size, void* d_ws, size_t ws_size,
                              hipStream_t stream)
{
  (void)in_sizes; (void)n_in; (void)out_size; (void)ws_size;
  const float* x   = (const float*)d_in[0];
  const float* pb  = (const float*)d_in[1];
  const float* W   = (const float*)d_in[2];   // [H, D] row-major
  const float* lb  = (const float*)d_in[3];
  const int*   st0 = (const int*)  d_in[4];

  const size_t BH = (size_t)BATCH * HID, BD = (size_t)BATCH * DIM;
  float* out     = (float*)d_out;
  float* lat_k   = out;
  float* xb      = lat_k  + BH;
  float* pre_m   = xb     + BD;    // pre, later masked in place -> pre_m2
  float* lat_4k  = pre_m  + BH;
  float* lat_ax  = lat_4k + BH;
  float* rec_k   = lat_ax + BH;
  float* rec_4k  = rec_k  + BD;
  float* rec_ax  = rec_4k + BD;
  float* rec_kpb = rec_ax + BD;
  int*   st_out  = (int*)(rec_kpb + BD);

  char* wp = (char*)d_ws;
  auto wsalloc = [&](size_t bytes) { char* p = wp; wp += (bytes + 255) & ~(size_t)255; return p; };
  float* v64   = (float*)wsalloc((size_t)BATCH * TOPK  * 4);
  int*   i64   = (int*)  wsalloc((size_t)BATCH * TOPK  * 4);
  float* v4k   = (float*)wsalloc((size_t)BATCH * TOPK4 * 4);
  int*   i4k   = (int*)  wsalloc((size_t)BATCH * TOPK4 * 4);
  float* vax   = (float*)wsalloc((size_t)BATCH * KAUX  * 4);
  int*   iax   = (int*)  wsalloc((size_t)BATCH * KAUX  * 4);
  int*   hits1 = (int*)  wsalloc(HID * 4);
  int*   stats1= (int*)  wsalloc(HID * 4);
  int*   hits2 = (int*)  wsalloc(HID * 4);

  const dim3 blk(256);

  // 1) xb = x - pre_bias
  ew_sub_colbias<<<4096, blk, 0, stream>>>(x, pb, xb, (long)BD, DIM);

  // 2) pre = (x - pre_bias) @ W^T + latent_bias   (W row-major == [N][K] tile)
  gemm_bf16x2<true, true><<<dim3(HID / BN, BATCH / BM), blk, 0, stream>>>(
      x, DIM, pb, W, DIM, lb, pre_m, HID, DIM);

  // 3) top-64 on pre
  topk_rows<TOPK><<<BATCH, blk, 0, stream>>>(pre_m, nullptr, v64, i64);

  // 4) stats1
  hipMemsetAsync(hits1, 0, HID * 4, stream);
  hits_accum<<<1024, blk, 0, stream>>>(v64, i64, (long)BATCH * TOPK, hits1);
  stats_update<<<HID / 256, blk, 0, stream>>>(hits1, st0, stats1);

  // 5) top-256 / top-512 on dead1-masked pre (mask applied on the fly)
  topk_rows<TOPK4><<<BATCH, blk, 0, stream>>>(pre_m, stats1, v4k, i4k);
  topk_rows<KAUX><<<BATCH, blk, 0, stream>>>(pre_m, stats1, vax, iax);

  // 6) stats2 from the 4k selection -> final stats output
  hipMemsetAsync(hits2, 0, HID * 4, stream);
  hits_accum<<<1024, blk, 0, stream>>>(v4k, i4k, (long)BATCH * TOPK4, hits2);
  stats_update<<<HID / 256, blk, 0, stream>>>(hits2, stats1, st_out);

  // 7) pre_m2 = pre * dead1 * dead2  (in place)
  mask_pre<<<8192, blk, 0, stream>>>(pre_m, stats1, st_out, (long)BH);

  // 8) dense relu latents
  hipMemsetAsync(lat_k,  0, BH * 4, stream);
  hipMemsetAsync(lat_4k, 0, BH * 4, stream);
  hipMemsetAsync(lat_ax, 0, BH * 4, stream);
  scatter_relu<<<BATCH, blk, 0, stream>>>(v64, i64, TOPK,  lat_k);
  scatter_relu<<<BATCH, blk, 0, stream>>>(v4k, i4k, TOPK4, lat_4k);
  scatter_relu<<<BATCH, blk, 0, stream>>>(vax, iax, KAUX,  lat_ax);

  // 9) decodes: lat @ W + pre_bias   (W as [K][N] tile)
  gemm_bf16x2<false, false><<<dim3(DIM / BN, BATCH / BM), blk, 0, stream>>>(
      lat_k, HID, nullptr, W, DIM, pb, rec_k, DIM, HID);
  gemm_bf16x2<false, false><<<dim3(DIM / BN, BATCH / BM), blk, 0, stream>>>(
      lat_4k, HID, nullptr, W, DIM, pb, rec_4k, DIM, HID);
  gemm_bf16x2<false, false><<<dim3(DIM / BN, BATCH / BM), blk, 0, stream>>>(
      lat_ax, HID, nullptr, W, DIM, pb, rec_ax, DIM, HID);

  // 10) recons_k - pre_bias
  ew_sub_colbias<<<4096, blk, 0, stream>>>(rec_k, pb, rec_kpb, (long)BD, DIM);
}
